// MultiHeadAttention_37495064494463
// MI455X (gfx1250) — matrix operand order repaired and checked by reading, not compiled, not run on hardware
//
#include <hip/hip_runtime.h>
#include <hip/hip_bf16.h>

#define D 1024
#define NH 16
#define HD 64
#define SEQ 2048
#define PAD 8
#define KB 64

typedef __attribute__((ext_vector_type(16))) __bf16 v16bf;
typedef __attribute__((ext_vector_type(8)))  __bf16 bf16x8;
typedef __attribute__((ext_vector_type(8)))  float  v8f;
typedef __attribute__((ext_vector_type(4)))  float  f32x4;

union FragAB { v16bf v; bf16x8 h[2]; };

__device__ __forceinline__ v8f wmma_bf16(v16bf a, v16bf b, v8f c) {
  return __builtin_amdgcn_wmma_f32_16x16x32_bf16(false, a, false, b, (short)0, c,
                                                 false, false);
}

// xor-permute within 16-lane groups; all lanes active so the tied "old"
// operand of v_permlane16_b32 is never observed -> plain asm, no v_mov.
__device__ __forceinline__ float pl16_xor(float x, unsigned s0, unsigned s1) {
  float d;
  asm("v_permlane16_b32 %0, %1, %2, %3" : "=v"(d) : "v"(x), "s"(s0), "s"(s1));
  return d;
}
__device__ __forceinline__ float redmax16(float x) {
  x = fmaxf(x, pl16_xor(x, 0x67452301u, 0xEFCDAB89u));  // xor 1
  x = fmaxf(x, pl16_xor(x, 0x45670123u, 0xCDEF89ABu));  // xor 3
  x = fmaxf(x, pl16_xor(x, 0x01234567u, 0x89ABCDEFu));  // xor 7
  x = fmaxf(x, pl16_xor(x, 0xFEDCBA98u, 0x76543210u));  // xor 8
  return x;
}

// ---------------------------------------------------------------------------
// GEMM: out = A[4096 x 1024] @ W[1024 x 1024]^T + bias.
// MODE 0: Q -> bf16 [B,H,S,HD] * 1/sqrt(HD);  MODE 1: K -> bf16 [B,H,S,HD]
// MODE 2: V -> bf16 [B,H,HD,S];               MODE 3: out -> f32 [M,1024]
// Block tile 128x128, 8 waves (4Mx2N), wave tile 32x64, K-step 32,
// double-buffered LDS with software-pipelined staging.
// ---------------------------------------------------------------------------
template<int MODE, typename AT>
__global__ __launch_bounds__(256)
void gemm_qkvo(const AT* __restrict__ A, const float* __restrict__ W,
               const float* __restrict__ bias, void* __restrict__ dstv)
{
  __shared__ __bf16 sA[2][128][32 + PAD];
  __shared__ __bf16 sB[2][128][32 + PAD];

  const int tid = threadIdx.x, lane = tid & 31, wid = tid >> 5;
  const int blockM = blockIdx.x * 128, blockN = blockIdx.y * 128;
  const int waveM = (wid >> 1) * 32, waveN = (wid & 1) * 64;

  const int srow = tid >> 1, scol = (tid & 1) * 16;

  f32x4 ra[4]; bf16x8 rabf[2]; f32x4 rb[4];

  auto loadStage = [&](int kt) {
    if constexpr (sizeof(AT) == 4) {
      const float* sa = (const float*)A + (size_t)(blockM + srow) * D + kt + scol;
      ra[0] = *(const f32x4*)(sa);
      ra[1] = *(const f32x4*)(sa + 4);
      ra[2] = *(const f32x4*)(sa + 8);
      ra[3] = *(const f32x4*)(sa + 12);
    } else {
      const __bf16* sa = (const __bf16*)A + (size_t)(blockM + srow) * D + kt + scol;
      rabf[0] = *(const bf16x8*)(sa);
      rabf[1] = *(const bf16x8*)(sa + 8);
    }
    const float* sb = W + (size_t)(blockN + srow) * D + kt + scol;
    rb[0] = *(const f32x4*)(sb);
    rb[1] = *(const f32x4*)(sb + 4);
    rb[2] = *(const f32x4*)(sb + 8);
    rb[3] = *(const f32x4*)(sb + 12);
  };
  auto storeStage = [&](int buf) {
    if constexpr (sizeof(AT) == 4) {
      bf16x8 y0, y1;
#pragma unroll
      for (int i = 0; i < 4; ++i) {
        y0[i] = (__bf16)ra[0][i]; y0[4 + i] = (__bf16)ra[1][i];
        y1[i] = (__bf16)ra[2][i]; y1[4 + i] = (__bf16)ra[3][i];
      }
      *(bf16x8*)&sA[buf][srow][scol]     = y0;
      *(bf16x8*)&sA[buf][srow][scol + 8] = y1;
    } else {
      *(bf16x8*)&sA[buf][srow][scol]     = rabf[0];
      *(bf16x8*)&sA[buf][srow][scol + 8] = rabf[1];
    }
    bf16x8 z0, z1;
#pragma unroll
    for (int i = 0; i < 4; ++i) {
      z0[i] = (__bf16)rb[0][i]; z0[4 + i] = (__bf16)rb[1][i];
      z1[i] = (__bf16)rb[2][i]; z1[4 + i] = (__bf16)rb[3][i];
    }
    *(bf16x8*)&sB[buf][srow][scol]     = z0;
    *(bf16x8*)&sB[buf][srow][scol + 8] = z1;
  };

  v8f acc[2][4] = {};

  loadStage(0);
  storeStage(0);
  const int NIT = D / 32;
  for (int it = 0; it < NIT; ++it) {
    const int cur = it & 1;
    __syncthreads();
    const bool more = (it + 1) < NIT;
    if (more) loadStage((it + 1) * 32);   // global loads issue early

    FragAB fa[2], fb[4];
    const int kA = (lane >> 4) * 8;
    const int kB2 = (lane >> 4) * 8;
#pragma unroll
    for (int mt = 0; mt < 2; ++mt) {
      const int row = waveM + mt * 16 + (lane & 15);
      fa[mt].h[0] = *(const bf16x8*)&sA[cur][row][kA];
      fa[mt].h[1] = *(const bf16x8*)&sA[cur][row][kA + 16];
    }
#pragma unroll
    for (int nt = 0; nt < 4; ++nt) {
      const int row = waveN + nt * 16 + (lane & 15);
      fb[nt].h[0] = *(const bf16x8*)&sB[cur][row][kB2];
      fb[nt].h[1] = *(const bf16x8*)&sB[cur][row][kB2 + 16];
    }
#pragma unroll
    for (int mt = 0; mt < 2; ++mt)
#pragma unroll
      for (int nt = 0; nt < 4; ++nt)
        acc[mt][nt] = wmma_bf16(fa[mt].v, fb[nt].v, acc[mt][nt]);

    if (more) storeStage(cur ^ 1);        // LDS stores after compute
  }

  // ---- epilogue ----
#pragma unroll
  for (int nt = 0; nt < 4; ++nt) {
    const int n_g = blockN + waveN + nt * 16 + (lane & 15);
    const float bn = bias[n_g];
#pragma unroll
    for (int mt = 0; mt < 2; ++mt) {
#pragma unroll
      for (int r = 0; r < 8; ++r) {
        const int m_g = blockM + waveM + mt * 16 + r + 8 * (lane >> 4);
        const float v = acc[mt][nt][r] + bn;
        const int b = m_g / SEQ, s = m_g % SEQ;
        const int h = n_g / HD,  hd = n_g % HD;
        if constexpr (MODE == 0) {
          ((__bf16*)dstv)[((size_t)(b * NH + h) * SEQ + s) * HD + hd] = (__bf16)(v * 0.125f);
        } else if constexpr (MODE == 1) {
          ((__bf16*)dstv)[((size_t)(b * NH + h) * SEQ + s) * HD + hd] = (__bf16)v;
        } else if constexpr (MODE == 2) {
          ((__bf16*)dstv)[((size_t)(b * NH + h) * HD + hd) * SEQ + s] = (__bf16)v;
        } else {
          ((float*)dstv)[(size_t)m_g * D + n_g] = v;
        }
      }
    }
  }
}

// ---------------------------------------------------------------------------
// Flash attention. Grid: (S/128, B*H). 8 waves x 16 queries, 64 keys/iter,
// double-buffered K/V tiles, permlane16 max tree, row-sums via WMMA vs ones.
// ---------------------------------------------------------------------------
__global__ __launch_bounds__(256)
void attn_kernel(const __bf16* __restrict__ Q, const __bf16* __restrict__ K,
                 const __bf16* __restrict__ Vt, __bf16* __restrict__ ctx)
{
  __shared__ __bf16 sK[2][KB][HD + PAD];
  __shared__ __bf16 sV[2][HD][KB + PAD];
  __shared__ __bf16 pbuf[8][16][KB + PAD];

  const int tid = threadIdx.x, lane = tid & 31, wid = tid >> 5;
  const int bh = blockIdx.y;           // 0..31
  const int b = bh >> 4, h = bh & 15;
  const int qBase = blockIdx.x * 128 + wid * 16;

  const __bf16* Kbase = K  + (size_t)bh * SEQ * HD;
  const __bf16* Vbase = Vt + (size_t)bh * HD * SEQ;

  const int srow = tid >> 2, sseg = (tid & 3) * 16;
  bf16x8 rk0, rk1, rv0, rv1;

  auto loadStage = [&](int kb) {
    const __bf16* ks = Kbase + (size_t)(kb + srow) * HD + sseg;
    rk0 = *(const bf16x8*)ks;
    rk1 = *(const bf16x8*)(ks + 8);
    const __bf16* vs = Vbase + (size_t)srow * SEQ + kb + sseg;
    rv0 = *(const bf16x8*)vs;
    rv1 = *(const bf16x8*)(vs + 8);
  };
  auto storeStage = [&](int buf) {
    *(bf16x8*)&sK[buf][srow][sseg]     = rk0;
    *(bf16x8*)&sK[buf][srow][sseg + 8] = rk1;
    *(bf16x8*)&sV[buf][srow][sseg]     = rv0;
    *(bf16x8*)&sV[buf][srow][sseg + 8] = rv1;
  };

  // Q fragments (A-layout), held for the whole key loop
  FragAB aq[2];
  {
    const int qs = qBase + (lane & 15);
    const __bf16* qrow = Q + ((size_t)bh * SEQ + qs) * HD;
#pragma unroll
    for (int dc = 0; dc < 2; ++dc) {
      const int d1 = dc * 32 + (lane >> 4) * 8;
      aq[dc].h[0] = *(const bf16x8*)(qrow + d1);
      aq[dc].h[1] = *(const bf16x8*)(qrow + d1 + 16);
    }
  }

  // constant all-ones B fragment: psum = P @ ones gives row sums of P
  FragAB onesf;
#pragma unroll
  for (int i = 0; i < 8; ++i) { onesf.h[0][i] = (__bf16)1.0f; onesf.h[1][i] = (__bf16)1.0f; }

  float mrow[8], lrow[8];
  v8f o[4] = {};
#pragma unroll
  for (int r = 0; r < 8; ++r) { mrow[r] = -1e30f; lrow[r] = 0.f; }

  loadStage(0);
  storeStage(0);
  const int NIT = SEQ / KB;
  for (int it = 0; it < NIT; ++it) {
    const int cur = it & 1;
    __syncthreads();
    const bool more = (it + 1) < NIT;
    if (more) loadStage((it + 1) * KB);   // prefetch next K/V tile

    // scores: 16 queries x 64 keys (4 tiles)
    v8f s[4] = {};
#pragma unroll
    for (int dc = 0; dc < 2; ++dc) {
      const int dbase = dc * 32 + (lane >> 4) * 8;
#pragma unroll
      for (int kt = 0; kt < 4; ++kt) {
        FragAB fk;
        const int key = kt * 16 + (lane & 15);
        fk.h[0] = *(const bf16x8*)&sK[cur][key][dbase];
        fk.h[1] = *(const bf16x8*)&sK[cur][key][dbase + 16];
        s[kt] = wmma_bf16(aq[dc].v, fk.v, s[kt]);
      }
    }

    // online softmax rescale (row max via permlane16 tree)
#pragma unroll
    for (int r = 0; r < 8; ++r) {
      float rm = fmaxf(fmaxf(s[0][r], s[1][r]), fmaxf(s[2][r], s[3][r]));
      rm = redmax16(rm);
      const float mn = fmaxf(mrow[r], rm);
      const float sc = __expf(mrow[r] - mn);
      mrow[r] = mn;
      lrow[r] *= sc;
#pragma unroll
      for (int kt = 0; kt < 4; ++kt) s[kt][r] = __expf(s[kt][r] - mn);
#pragma unroll
      for (int nt = 0; nt < 4; ++nt) o[nt][r] *= sc;
    }

    // transpose P: C-layout -> A-layout via wave-private LDS bounce
#pragma unroll
    for (int r = 0; r < 8; ++r) {
      const int prow = r + 8 * (lane >> 4);
#pragma unroll
      for (int kt = 0; kt < 4; ++kt)
        pbuf[wid][prow][kt * 16 + (lane & 15)] = (__bf16)s[kt][r];
    }
    asm volatile("s_wait_dscnt 0" ::: "memory");  // same-wave DS in-order drain

    v8f psum = {};
#pragma unroll
    for (int dc = 0; dc < 2; ++dc) {
      FragAB fp;
      const int prow = lane & 15;
      const int kc = dc * 32 + (lane >> 4) * 8;
      fp.h[0] = *(const bf16x8*)&pbuf[wid][prow][kc];
      fp.h[1] = *(const bf16x8*)&pbuf[wid][prow][kc + 16];
      psum = wmma_bf16(fp.v, onesf.v, psum);   // row sums of P on the matrix pipe
#pragma unroll
      for (int nt = 0; nt < 4; ++nt) {
        FragAB fv;
        const int n = nt * 16 + (lane & 15);
        const int kv = dc * 32 + (lane >> 4) * 8;
        fv.h[0] = *(const bf16x8*)&sV[cur][n][kv];
        fv.h[1] = *(const bf16x8*)&sV[cur][n][kv + 16];
        o[nt] = wmma_bf16(fp.v, fv.v, o[nt]);
      }
    }
#pragma unroll
    for (int r = 0; r < 8; ++r) lrow[r] += psum[r];

    if (more) storeStage(cur ^ 1);        // LDS stores after compute
  }

  // normalize and write ctx [B, S, D] as bf16
#pragma unroll
  for (int r = 0; r < 8; ++r) {
    const float inv = 1.0f / lrow[r];
    const int qs = qBase + r + 8 * (lane >> 4);
    const size_t base = ((size_t)b * SEQ + qs) * D + h * HD;
#pragma unroll
    for (int nt = 0; nt < 4; ++nt)
      ctx[base + nt * 16 + (lane & 15)] = (__bf16)(o[nt][r] * inv);
  }
}

// ---------------------------------------------------------------------------
extern "C" void kernel_launch(void* const* d_in, const int* in_sizes, int n_in,
                              void* d_out, int out_size, void* d_ws, size_t ws_size,
                              hipStream_t stream) {
  const float* x  = (const float*)d_in[0];
  const float* Wq = (const float*)d_in[1];
  const float* bq = (const float*)d_in[2];
  const float* Wk = (const float*)d_in[3];
  const float* bk = (const float*)d_in[4];
  const float* Wv = (const float*)d_in[5];
  const float* bv = (const float*)d_in[6];
  const float* Wo = (const float*)d_in[7];
  const float* bo = (const float*)d_in[8];

  char* ws = (char*)d_ws;
  const size_t MB8 = (size_t)8 * 1024 * 1024;
  __bf16* qb  = (__bf16*)(ws);            // Q  [B,H,S,HD]  8 MB
  __bf16* kb  = (__bf16*)(ws + MB8);      // K  [B,H,S,HD]  8 MB
  __bf16* vtb = (__bf16*)(ws + 2 * MB8);  // Vt [B,H,HD,S]  8 MB
  __bf16* cb  = (__bf16*)(ws + 3 * MB8);  // ctx [B,S,D]    8 MB

  dim3 g(32, 8), blk(256);
  gemm_qkvo<0, float ><<<g, blk, 0, stream>>>(x,  Wq, bq, qb);
  gemm_qkvo<1, float ><<<g, blk, 0, stream>>>(x,  Wk, bk, kb);
  gemm_qkvo<2, float ><<<g, blk, 0, stream>>>(x,  Wv, bv, vtb);
  attn_kernel<<<dim3(16, 32), blk, 0, stream>>>(qb, kb, vtb, cb);
  gemm_qkvo<3, __bf16><<<g, blk, 0, stream>>>(cb, Wo, bo, (void*)d_out);
}
